// Draft_Attention_8160437862549
// MI455X (gfx1250) — compile-verified, matrix operand order checked by reading, not run on hardware
//
#include <hip/hip_runtime.h>

#define L_TOK   7680
#define NHEAD   16
#define HDIM    64
#define S_POOL  60
#define NB      60          // key/query blocks in reorganized sequence
#define MB      128         // tokens per block
#define TOKSTRIDE (NHEAD*HDIM)   // 1024 floats per token row

typedef __attribute__((ext_vector_type(16))) __bf16 v16bf;
typedef __attribute__((ext_vector_type(8)))  __bf16 v8bf;
typedef __attribute__((ext_vector_type(8)))  float  v8f;

// reorganized position j -> original token index
__device__ __forceinline__ int reorg_orig(int j) {
  int p  = j / 640;
  int r  = j - p * 640;
  int c  = r >> 7;      // sub-block column (0..4)
  int r2 = r & 127;
  int b  = r2 >> 4;     // block row (0..7)
  int i  = r2 & 15;
  return p * 640 + b * 80 + c * 16 + i;
}

// ---- xor-butterfly reductions within 16-lane halves -----------------------
// Prefer v_permlane16_b32 (pure VALU) over ds_bpermute (LDS pipeline).
#if __has_builtin(__builtin_amdgcn_permlane16)
__device__ __forceinline__ float pl16(float v, unsigned s0, unsigned s1) {
  unsigned u = __float_as_uint(v);
  return __uint_as_float(__builtin_amdgcn_permlane16(u, u, s0, s1, false, false));
}
__device__ __forceinline__ float red16_max(float v) {
  v = fmaxf(v, pl16(v, 0x67452301u, 0xEFCDAB89u));   // xor 1
  v = fmaxf(v, pl16(v, 0x54761032u, 0xDCFE98BAu));   // xor 2
  v = fmaxf(v, pl16(v, 0x32107654u, 0xBA98FEDCu));   // xor 4
  v = fmaxf(v, pl16(v, 0xFEDCBA98u, 0x76543210u));   // xor 8
  return v;
}
__device__ __forceinline__ float red16_sum(float v) {
  v += pl16(v, 0x67452301u, 0xEFCDAB89u);
  v += pl16(v, 0x54761032u, 0xDCFE98BAu);
  v += pl16(v, 0x32107654u, 0xBA98FEDCu);
  v += pl16(v, 0xFEDCBA98u, 0x76543210u);
  return v;
}
#else
__device__ __forceinline__ float red16_max(float v) {
  v = fmaxf(v, __shfl_xor(v, 1, 32));
  v = fmaxf(v, __shfl_xor(v, 2, 32));
  v = fmaxf(v, __shfl_xor(v, 4, 32));
  v = fmaxf(v, __shfl_xor(v, 8, 32));
  return v;
}
__device__ __forceinline__ float red16_sum(float v) {
  v += __shfl_xor(v, 1, 32);
  v += __shfl_xor(v, 2, 32);
  v += __shfl_xor(v, 4, 32);
  v += __shfl_xor(v, 8, 32);
  return v;
}
#endif

__device__ __forceinline__ v16bf cat16(v8bf a, v8bf b) {
  return __builtin_shufflevector(a, b, 0,1,2,3,4,5,6,7,8,9,10,11,12,13,14,15);
}

// load 8 consecutive f32, scale, convert to bf16, place at d[off..off+7]
__device__ __forceinline__ void load8cvt(const float* __restrict__ p, float s, v16bf& d, int off) {
  const float4* p4 = reinterpret_cast<const float4*>(p);
  float4 x = p4[0], y = p4[1];
  d[off+0] = (__bf16)(x.x*s); d[off+1] = (__bf16)(x.y*s);
  d[off+2] = (__bf16)(x.z*s); d[off+3] = (__bf16)(x.w*s);
  d[off+4] = (__bf16)(y.x*s); d[off+5] = (__bf16)(y.y*s);
  d[off+6] = (__bf16)(y.z*s); d[off+7] = (__bf16)(y.w*s);
}

// ---------------- Phase 1: 8x16 average pooling of q and k -----------------
__global__ void pool_kernel(const float* __restrict__ q, const float* __restrict__ k,
                            float* __restrict__ qs, float* __restrict__ ks) {
  int s = blockIdx.x;          // pooled token 0..59
  int h = blockIdx.y;          // head
  int which = blockIdx.z;      // 0=q, 1=k
  int d = threadIdx.x;         // dim 0..63
  const float* src = which ? k : q;
  float* dst = which ? ks : qs;
  int f = s / 30, rem = s % 30, yy = rem / 5, xx = rem % 5;
  int base = f * 3840 + yy * 8 * 80 + xx * 16;
  float acc = 0.0f;
  for (int py = 0; py < 8; ++py) {
    const float* row = src + (long)(base + py * 80) * TOKSTRIDE + h * HDIM + d;
    for (int px = 0; px < 16; ++px) acc += row[(long)px * TOKSTRIDE];
  }
  dst[((long)h * S_POOL + s) * HDIM + d] = acc * (1.0f / 128.0f);
}

// ---------------- Phase 2: draft attention map (softmax rows) ---------------
__global__ void scores_kernel(const float* __restrict__ qs, const float* __restrict__ ks,
                              float* __restrict__ attn) {
  int i = blockIdx.x;          // pooled query row
  int h = blockIdx.y;
  int t = threadIdx.x;         // 64 threads; t<60 = pooled key col
  __shared__ float sc[S_POOL];
  __shared__ float mx, sm;
  if (t < S_POOL) {
    const float* qp = qs + ((long)h * S_POOL + i) * HDIM;
    const float* kp = ks + ((long)h * S_POOL + t) * HDIM;
    float d = 0.0f;
    #pragma unroll
    for (int e = 0; e < HDIM; ++e) d += qp[e] * kp[e];
    sc[t] = d * 0.125f;        // 1/sqrt(64)
  }
  __syncthreads();
  if (t == 0) {
    float m = sc[0];
    for (int e = 1; e < S_POOL; ++e) m = fmaxf(m, sc[e]);
    mx = m;
  }
  __syncthreads();
  if (t < S_POOL) sc[t] = __expf(sc[t] - mx);
  __syncthreads();
  if (t == 0) {
    float s = 0.0f;
    for (int e = 0; e < S_POOL; ++e) s += sc[e];
    sm = s;
  }
  __syncthreads();
  if (t < S_POOL) attn[(long)h * S_POOL * S_POOL + (long)i * S_POOL + t] = sc[t] / sm;
}

// ------- Phase 3: exact per-head percentile threshold + block mask ----------
__global__ void thresh_kernel(const float* __restrict__ attn, unsigned char* __restrict__ mask) {
  const int N = S_POOL * S_POOL;   // 3600
  const int KK = 3239;             // index of sorted[kk-1], kk = int(0.9*3600)
  int h = blockIdx.x, tid = threadIdx.x;
  __shared__ float a[S_POOL * S_POOL];
  __shared__ float thr;
  for (int idx = tid; idx < N; idx += 256) a[idx] = attn[(long)h * N + idx];
  __syncthreads();
  for (int idx = tid; idx < N; idx += 256) {
    float v = a[idx];
    int cl = 0, cle = 0;
    for (int j = 0; j < N; ++j) { cl += (a[j] < v); cle += (a[j] <= v); }
    if (cl <= KK && KK < cle) thr = v;   // all writers write the same value
  }
  __syncthreads();
  float t = thr;
  for (int idx = tid; idx < N; idx += 256)
    mask[(long)h * N + idx] = (a[idx] >= t) ? 1 : 0;
}

// ------- Phase 4a: K -> bf16, reorganized, row-major: kbf[h][j][d] ----------
__global__ void kbf_kernel(const float* __restrict__ k, __bf16* __restrict__ kbf) {
  int idx = blockIdx.x * 256 + threadIdx.x;   // over 16*7680*64, d fastest
  int d = idx & 63;
  int t = idx >> 6;
  int j = t % L_TOK;
  int h = t / L_TOK;
  kbf[idx] = (__bf16)k[(long)reorg_orig(j) * TOKSTRIDE + h * HDIM + d];
}

// ------- Phase 4b: V -> bf16, transposed + reorganized: vt[h][d][j] ---------
__global__ void vt_kernel(const float* __restrict__ v, __bf16* __restrict__ vt) {
  int idx = blockIdx.x * 256 + threadIdx.x;   // over 16*64*7680, j fastest
  int j  = idx % L_TOK;
  int hd = idx / L_TOK;
  int d  = hd & 63, h = hd >> 6;
  vt[idx] = (__bf16)v[(long)reorg_orig(j) * TOKSTRIDE + h * HDIM + d];
}

// ------- Phase 5: block-sparse flash attention (WMMA bf16, f32 accum) -------
__global__ __launch_bounds__(256) void draft_attn_kernel(
    const float* __restrict__ q, const __bf16* __restrict__ kbf,
    const __bf16* __restrict__ vt, const unsigned char* __restrict__ mask,
    float* __restrict__ out)
{
  const int qb   = blockIdx.x;   // query block 0..59 (128 rows each)
  const int h    = blockIdx.y;   // head
  const int tid  = threadIdx.x;
  const int w    = tid >> 5;     // wave 0..7, owns 16 query rows
  const int lane = tid & 31;
  const int col  = lane & 15;    // N / M-within-tile index
  const int kh   = lane >> 4;    // K-half (A/B layout) == row-half (C layout)

  __shared__ unsigned char smask[64];
  __shared__ __align__(16) __bf16 pstage[8][16 * 128];  // per-wave P staging (4 KB)

  if (tid < NB) smask[tid] = mask[(long)h * NB * NB + qb * NB + tid];
  __syncthreads();

  // Q A-tiles (16x32 bf16): dims 0..31 / 32..63; 1/sqrt(D) folded into cvt
  const int qrow = qb * MB + w * 16 + col;
  const float* qp = q + (long)reorg_orig(qrow) * TOKSTRIDE + h * HDIM;
  v16bf aq0, aq1;
  load8cvt(qp +      kh * 8, 0.125f, aq0, 0);
  load8cvt(qp + 16 + kh * 8, 0.125f, aq0, 8);
  load8cvt(qp + 32 + kh * 8, 0.125f, aq1, 0);
  load8cvt(qp + 48 + kh * 8, 0.125f, aq1, 8);

  v8f oacc[4] = {};              // 16x64 f32 output accumulator (4 C tiles)
  float mrow[8], lrow[8];
  #pragma unroll
  for (int r = 0; r < 8; ++r) { mrow[r] = -1e30f; lrow[r] = 0.0f; }

  int nblk = 0;
  #pragma unroll 1
  for (int kb = 0; kb < NB; ++kb) {
    if (!smask[kb]) continue;    // block-sparse skip (exact: exp(-inf)=0)
    ++nblk;

    // ---- S = (Q/8) K^T for 16 rows x 128 keys: 8 C tiles, D=64 chained ----
    v8f sc[8];
    #pragma unroll
    for (int nt = 0; nt < 8; ++nt) {
      const __bf16* kp = kbf + ((long)h * L_TOK + kb * MB + nt * 16 + col) * HDIM;
      v8bf l0 = *reinterpret_cast<const v8bf*>(kp +      kh * 8);
      v8bf h0 = *reinterpret_cast<const v8bf*>(kp + 16 + kh * 8);
      v8f acc = {};
      acc = __builtin_amdgcn_wmma_f32_16x16x32_bf16(false, aq0, false, cat16(l0, h0), (short)0, acc, false, false);
      v8bf l1 = *reinterpret_cast<const v8bf*>(kp + 32 + kh * 8);
      v8bf h1 = *reinterpret_cast<const v8bf*>(kp + 48 + kh * 8);
      acc = __builtin_amdgcn_wmma_f32_16x16x32_bf16(false, aq1, false, cat16(l1, h1), (short)0, acc, false, false);
      sc[nt] = acc;
    }

    // ---- one online-softmax update per 128-key block ----
    #pragma unroll
    for (int r = 0; r < 8; ++r) {          // C layout: VGPR r = row r + 8*kh
      float mloc = sc[0][r];
      #pragma unroll
      for (int nt = 1; nt < 8; ++nt) mloc = fmaxf(mloc, sc[nt][r]);
      float rm = red16_max(mloc);
      float mn = fmaxf(mrow[r], rm);
      float scl = __expf(mrow[r] - mn);
      mrow[r] = mn;
      float sum = 0.0f;
      int row = r + 8 * kh;
      #pragma unroll
      for (int nt = 0; nt < 8; ++nt) {
        float p = __expf(sc[nt][r] - mn);
        sum += p;
        pstage[w][row * 128 + nt * 16 + col] = (__bf16)p;   // row-major P
      }
      float rs = red16_sum(sum);
      lrow[r] = lrow[r] * scl + rs;
      oacc[0][r] *= scl; oacc[1][r] *= scl; oacc[2][r] *= scl; oacc[3][r] *= scl;
    }

    // ---- O += P V over 4 chunks of 32 keys (in-order DS within wave) ----
    #pragma unroll
    for (int ch = 0; ch < 4; ++ch) {
      v8bf plo = *reinterpret_cast<const v8bf*>(&pstage[w][col * 128 + ch * 32 +      kh * 8]);
      v8bf phi = *reinterpret_cast<const v8bf*>(&pstage[w][col * 128 + ch * 32 + 16 + kh * 8]);
      v16bf pa = cat16(plo, phi);
      const int kbase = kb * MB + ch * 32;
      #pragma unroll
      for (int t = 0; t < 4; ++t) {
        const __bf16* vp = vt + ((long)(h * HDIM + t * 16 + col)) * L_TOK + kbase;
        v8bf b0 = *reinterpret_cast<const v8bf*>(vp +      kh * 8);
        v8bf b1 = *reinterpret_cast<const v8bf*>(vp + 16 + kh * 8);
        oacc[t] = __builtin_amdgcn_wmma_f32_16x16x32_bf16(false, pa, false, cat16(b0, b1), (short)0, oacc[t], false, false);
      }
    }
  }

  // ---- epilogue: divide by row-sum, scatter back through inverse perm ----
  #pragma unroll
  for (int r = 0; r < 8; ++r) {
    float inv = (nblk == 0) ? 0.0f : 1.0f / fmaxf(lrow[r], 1e-30f);
    int j = qb * MB + w * 16 + r + 8 * kh;
    float* op = out + (long)reorg_orig(j) * TOKSTRIDE + h * HDIM;
    op[col]      = oacc[0][r] * inv;
    op[col + 16] = oacc[1][r] * inv;
    op[col + 32] = oacc[2][r] * inv;
    op[col + 48] = oacc[3][r] * inv;
  }
}

extern "C" void kernel_launch(void* const* d_in, const int* in_sizes, int n_in,
                              void* d_out, int out_size, void* d_ws, size_t ws_size,
                              hipStream_t stream) {
  const float* q = (const float*)d_in[0];
  const float* k = (const float*)d_in[1];
  const float* v = (const float*)d_in[2];
  float* out = (float*)d_out;

  char* ws = (char*)d_ws;
  // workspace layout (~32 MB total)
  constexpr size_t BF_BYTES = (size_t)NHEAD * HDIM * L_TOK * 2;   // 15,728,640
  __bf16* vt   = (__bf16*)ws;
  __bf16* kbf  = (__bf16*)(ws + BF_BYTES);
  float*  qs   = (float*)(ws + 2 * BF_BYTES);
  float*  ks   = qs + (size_t)NHEAD * S_POOL * HDIM;
  float*  attn = ks + (size_t)NHEAD * S_POOL * HDIM;
  unsigned char* mask = (unsigned char*)(attn + (size_t)NHEAD * S_POOL * S_POOL);

  pool_kernel  <<<dim3(S_POOL, NHEAD, 2), 64,  0, stream>>>(q, k, qs, ks);
  scores_kernel<<<dim3(S_POOL, NHEAD),    64,  0, stream>>>(qs, ks, attn);
  thresh_kernel<<<NHEAD,                  256, 0, stream>>>(attn, mask);
  kbf_kernel   <<<(NHEAD * HDIM * L_TOK) / 256, 256, 0, stream>>>(k, kbf);
  vt_kernel    <<<(NHEAD * HDIM * L_TOK) / 256, 256, 0, stream>>>(v, vt);
  draft_attn_kernel<<<dim3(NB, NHEAD),    256, 0, stream>>>(q, kbf, vt, mask, out);
}
